// MambaPlusPlus_regular_53936199303396
// MI455X (gfx1250) — compile-verified, matrix operand order checked by reading, not run on hardware
//
#include <hip/hip_runtime.h>
#include <hip/hip_bf16.h>
#include <cstdint>
#include <cstddef>

// ---------- types ----------
typedef __attribute__((ext_vector_type(16))) __bf16 v16bf;
typedef __attribute__((ext_vector_type(8)))  __bf16 v8bf;
typedef __attribute__((ext_vector_type(4)))  __bf16 v4bf;
typedef __attribute__((ext_vector_type(8)))  float  v8f;
typedef __attribute__((ext_vector_type(4)))  float  v4f;

#define GAS __attribute__((address_space(1)))   // global address space

#define NROWS   8192      // B*L
#define EDIM    1024
#define DDIM    128
#define HHEADS  8
#define LSEQ    2048
#define VOCAB   32000
#define NGATES  2176      // 1024 (a) + 1024 (b) + 128 (dxt)

// ---------- epilogue ids ----------
#define EPI_GATES    0
#define EPI_RES_F32  1
#define EPI_GELU_B16 2
#define EPI_RES_B16  3
#define EPI_B16      4
#define EPI_F32      5

// =====================================================================
// Weight prep: combined gates weight [2176 x 1024] bf16, transposed
// =====================================================================
__global__ void build_gates_w(const float* __restrict__ Wa,
                              const float* __restrict__ Wb,
                              const float* __restrict__ Wd,
                              __bf16* __restrict__ dst) {
    int i = blockIdx.x * blockDim.x + threadIdx.x;
    if (i >= NGATES * EDIM) return;
    int j = i / EDIM;
    int e = i - j * EDIM;
    float v;
    if (j < 1024) {
        int h = j >> 7, d = j & 127;
        v = Wa[(size_t)h * EDIM * DDIM + (size_t)e * DDIM + d];
    } else if (j < 2048) {
        int jj = j - 1024, h = jj >> 7, d = jj & 127;
        v = Wb[(size_t)h * EDIM * DDIM + (size_t)e * DDIM + d];
    } else {
        v = Wd[(size_t)e * DDIM + (j - 2048)];
    }
    dst[(size_t)j * EDIM + e] = (__bf16)v;
}

// dst[n*K+k] = (bf16)src[k*N+n], batched on grid.y
__global__ void transpose_conv(const float* __restrict__ src,
                               __bf16* __restrict__ dst, int K, int N) {
    int z = blockIdx.y;
    src += (size_t)z * K * N;
    dst += (size_t)z * K * N;
    int i = blockIdx.x * blockDim.x + threadIdx.x;
    if (i >= K * N) return;
    int n = i / K, k = i - n * K;
    dst[(size_t)n * K + k] = (__bf16)src[(size_t)k * N + n];
}

// =====================================================================
// Embedding lookup with padding_idx = 0, emits bf16 [NROWS x EDIM]
// =====================================================================
__global__ void embed_k(const int* __restrict__ x,
                        const float* __restrict__ tab,
                        __bf16* __restrict__ emb) {
    int i = blockIdx.x * blockDim.x + threadIdx.x;   // one per 8 elems
    if (i >= NROWS * (EDIM / 8)) return;
    int n  = i >> 7;
    int e0 = (i & 127) << 3;
    int t  = x[n];
    v8bf out;
    if (t == 0) {
        for (int k = 0; k < 8; ++k) out[k] = (__bf16)0.0f;
    } else {
        const v4f GAS* p = (const v4f GAS*)(tab + (size_t)t * EDIM + e0);
        v4f a = p[0];
        v4f b = p[1];
        out[0]=(__bf16)a.x; out[1]=(__bf16)a.y; out[2]=(__bf16)a.z; out[3]=(__bf16)a.w;
        out[4]=(__bf16)b.x; out[5]=(__bf16)b.y; out[6]=(__bf16)b.z; out[7]=(__bf16)b.w;
    }
    *(v8bf GAS*)(emb + (size_t)n * EDIM + e0) = out;
}

// =====================================================================
// Generic bf16 WMMA GEMM: C[M,N] = A[M,K](row-major) x Bt[N,K]^T
// per-wave tile: 64 rows x 32 cols (4x2 of 16x16), K step 32.
// All streaming pointers forced to address_space(1) -> global_load/store.
// =====================================================================
struct GemmP {
    const __bf16* A;  int lda;
    const __bf16* Bt; int ldb;
    const float*  bias;
    float*  outf; __bf16* outb; int ldc;
    const float*  res; int ldres;
    float* ga; float* gb; float* gd;           // gates epilogue
    const float* ba; const float* bbv; const float* bd;
    int M, N, K;
};

__device__ __forceinline__ v16bf combine8(v8bf lo, v8bf hi) {
    return __builtin_shufflevector(lo, hi, 0,1,2,3,4,5,6,7,8,9,10,11,12,13,14,15);
}

__device__ __forceinline__ v16bf load_frag(const __bf16 GAS* p) {
    v8bf lo = *(const v8bf GAS*)p;
    v8bf hi = *(const v8bf GAS*)(p + 16);
    return combine8(lo, hi);
}

template <int EPI>
__global__ void __launch_bounds__(256, 1) gemm_wmma(GemmP p) {
    int wid = (int)((blockIdx.x * blockDim.x + threadIdx.x) >> 5);
    int mg = p.M >> 6;          // 64-row groups
    int ng = p.N >> 5;          // 32-col groups
    if (wid >= mg * ng) return;
    int m0 = (wid % mg) << 6;
    int n0 = (wid / mg) << 5;

    int lane  = threadIdx.x & 31;
    int lrow  = lane & 15;                 // fragment row (M%16 / N%16)
    int lkoff = (lane >> 4) << 3;          // K offset 0 or 8

    // per-lane fragment pointers (global AS), bumped by 32 per k-step
    const __bf16 GAS* ap[4];
    #pragma unroll
    for (int i = 0; i < 4; ++i)
        ap[i] = (const __bf16 GAS*)(p.A + (size_t)(m0 + 16 * i + lrow) * p.lda + lkoff);
    const __bf16 GAS* bp[2];
    #pragma unroll
    for (int j = 0; j < 2; ++j)
        bp[j] = (const __bf16 GAS*)(p.Bt + (size_t)(n0 + 16 * j + lrow) * p.ldb + lkoff);

    v8f acc[4][2];
    #pragma unroll
    for (int i = 0; i < 4; ++i)
        #pragma unroll
        for (int j = 0; j < 2; ++j)
            acc[i][j] = (v8f)(0.0f);

    for (int kk = 0; kk < p.K; kk += 32) {
        v16bf a[4];
        #pragma unroll
        for (int i = 0; i < 4; ++i) {
            a[i] = load_frag(ap[i]);
            ap[i] += 32;
        }
        #pragma unroll
        for (int j = 0; j < 2; ++j) {
            v16bf b = load_frag(bp[j]);
            bp[j] += 32;
            #pragma unroll
            for (int i = 0; i < 4; ++i)
                acc[i][j] = __builtin_amdgcn_wmma_f32_16x16x32_bf16(
                    false, a[i], false, b, (short)0, acc[i][j], false, false);
        }
    }

    // global-AS epilogue pointers
    float  GAS* outf = (float  GAS*)p.outf;
    __bf16 GAS* outb = (__bf16 GAS*)p.outb;
    const float GAS* res = (const float GAS*)p.res;
    float GAS* ga = (float GAS*)p.ga;
    float GAS* gb = (float GAS*)p.gb;
    float GAS* gd = (float GAS*)p.gd;

    int cn  = lane & 15;
    int rhi = (lane >> 4) << 3;            // +8 rows for upper half-wave
    #pragma unroll
    for (int i = 0; i < 4; ++i) {
        #pragma unroll
        for (int j = 0; j < 2; ++j) {
            int colb = n0 + 16 * j + cn;
            #pragma unroll
            for (int r = 0; r < 8; ++r) {
                int row = m0 + 16 * i + rhi + r;
                float v = acc[i][j][r];
                if (EPI == EPI_GATES) {
                    if (colb < 1024) {
                        ga[(size_t)row * 1024 + colb] = tanhf(v + p.ba[colb]);
                    } else if (colb < 2048) {
                        int c = colb - 1024;
                        gb[(size_t)row * 1024 + c] = v + p.bbv[c];
                    } else {
                        int c = colb - 2048;
                        gd[(size_t)row * 128 + c] = v + p.bd[c];
                    }
                } else if (EPI == EPI_RES_F32) {
                    outf[(size_t)row * p.ldc + colb] =
                        v + p.bias[colb] + res[(size_t)row * p.ldres + colb];
                } else if (EPI == EPI_GELU_B16) {
                    float x = v + p.bias[colb];
                    float g = 0.5f * x * (1.0f + erff(x * 0.70710678118654752f));
                    outb[(size_t)row * p.ldc + colb] = (__bf16)g;
                } else if (EPI == EPI_RES_B16) {
                    float x = v + p.bias[colb] + res[(size_t)row * p.ldres + colb];
                    outb[(size_t)row * p.ldc + colb] = (__bf16)x;
                } else if (EPI == EPI_B16) {
                    outb[(size_t)row * p.ldc + colb] = (__bf16)(v + p.bias[colb]);
                } else { // EPI_F32
                    outf[(size_t)row * p.ldc + colb] = v + p.bias[colb];
                }
            }
        }
    }
}

// =====================================================================
// Sequential scan: h_t = a_t * h_{t-1} + b_t, B*H*D = 4096 lanes
// =====================================================================
__global__ void scan_k(const float* __restrict__ a_buf,
                       const float* __restrict__ b_buf,
                       __bf16* __restrict__ hs) {
    int tid = blockIdx.x * blockDim.x + threadIdx.x;   // 0..4095
    int b = tid >> 10;
    int j = tid & 1023;
    size_t off = ((size_t)b * LSEQ) * 1024 + j;
    const float GAS* ap = (const float GAS*)(a_buf + off);
    const float GAS* bp = (const float GAS*)(b_buf + off);
    __bf16 GAS* hp = (__bf16 GAS*)(hs + off);
    float h = 0.0f;
    #pragma unroll 4
    for (int t = 0; t < LSEQ; ++t) {
        float av = *ap;
        float bv = *bp;
        h = fmaf(av, h, bv);
        *hp = (__bf16)h;
        ap += 1024; bp += 1024; hp += 1024;
    }
}

// =====================================================================
// Fuse: per-head LN of cout, weighted sum, z, LN(z), u = z + LN(z)
// One wave per token row; lane handles 4 contiguous dims.
// =====================================================================
__global__ void fuse_z_u(const float* __restrict__ cout,
                         const float* __restrict__ dxt,
                         const float* __restrict__ hw,
                         const float* __restrict__ hg,
                         const float* __restrict__ hb,
                         const float* __restrict__ ng,
                         const float* __restrict__ nb,
                         float* __restrict__ u_f,
                         __bf16* __restrict__ u_b) {
    int wave = (int)((blockIdx.x * blockDim.x + threadIdx.x) >> 5);
    if (wave >= NROWS) return;
    int lane = threadIdx.x & 31;
    int c0 = lane << 2;
    const float GAS* crow = (const float GAS*)(cout + (size_t)wave * 1024);

    float z0 = 0.f, z1 = 0.f, z2 = 0.f, z3 = 0.f;
    for (int h = 0; h < HHEADS; ++h) {
        v4f v = *(const v4f GAS*)(crow + h * 128 + c0);
        float s = v.x + v.y + v.z + v.w;
        float q = v.x*v.x + v.y*v.y + v.z*v.z + v.w*v.w;
        for (int m = 16; m > 0; m >>= 1) {
            s += __shfl_xor(s, m, 32);
            q += __shfl_xor(q, m, 32);
        }
        float mu = s * (1.0f / 128.0f);
        float rs = rsqrtf(q * (1.0f / 128.0f) - mu * mu + 1e-5f);
        float w  = hw[h];
        v4f g  = *(const v4f GAS*)((const float GAS*)hg + h * 128 + c0);
        v4f be = *(const v4f GAS*)((const float GAS*)hb + h * 128 + c0);
        z0 += w * ((v.x - mu) * rs * g.x + be.x);
        z1 += w * ((v.y - mu) * rs * g.y + be.y);
        z2 += w * ((v.z - mu) * rs * g.z + be.z);
        z3 += w * ((v.w - mu) * rs * g.w + be.w);
    }
    v4f dx = *(const v4f GAS*)((const float GAS*)dxt + (size_t)wave * 128 + c0);
    z0 = (dx.x + z0) * 0.125f;
    z1 = (dx.y + z1) * 0.125f;
    z2 = (dx.z + z2) * 0.125f;
    z3 = (dx.w + z3) * 0.125f;

    float s = z0 + z1 + z2 + z3;
    float q = z0*z0 + z1*z1 + z2*z2 + z3*z3;
    for (int m = 16; m > 0; m >>= 1) {
        s += __shfl_xor(s, m, 32);
        q += __shfl_xor(q, m, 32);
    }
    float mu = s * (1.0f / 128.0f);
    float rs = rsqrtf(q * (1.0f / 128.0f) - mu * mu + 1e-5f);
    v4f gv = *(const v4f GAS*)((const float GAS*)ng + c0);
    v4f bv = *(const v4f GAS*)((const float GAS*)nb + c0);
    float u0 = z0 + (z0 - mu) * rs * gv.x + bv.x;
    float u1 = z1 + (z1 - mu) * rs * gv.y + bv.y;
    float u2 = z2 + (z2 - mu) * rs * gv.z + bv.z;
    float u3 = z3 + (z3 - mu) * rs * gv.w + bv.w;

    v4f uf; uf.x = u0; uf.y = u1; uf.z = u2; uf.w = u3;
    *(v4f GAS*)((float GAS*)u_f + (size_t)wave * 128 + c0) = uf;
    v4bf ub; ub[0]=(__bf16)u0; ub[1]=(__bf16)u1; ub[2]=(__bf16)u2; ub[3]=(__bf16)u3;
    *(v4bf GAS*)((__bf16 GAS*)u_b + (size_t)wave * 128 + c0) = ub;
}

// =====================================================================
// host side
// =====================================================================
static inline int gemm_blocks(int M, int N) {
    int waves = (M >> 6) * (N >> 5);
    return (waves + 7) / 8;            // 8 waves (256 threads) per block
}

extern "C" void kernel_launch(void* const* d_in, const int* in_sizes, int n_in,
                              void* d_out, int out_size, void* d_ws, size_t ws_size,
                              hipStream_t stream) {
    (void)in_sizes; (void)n_in; (void)out_size; (void)ws_size;
    const int*   x        = (const int*)  d_in[0];
    const float* emb_tab  = (const float*)d_in[1];
    const float* Wa       = (const float*)d_in[2];
    const float* ba       = (const float*)d_in[3];
    const float* Wb       = (const float*)d_in[4];
    const float* bbv      = (const float*)d_in[5];
    const float* Wc       = (const float*)d_in[6];
    const float* bc       = (const float*)d_in[7];
    const float* hw       = (const float*)d_in[8];
    const float* hg       = (const float*)d_in[9];
    const float* hbeta    = (const float*)d_in[10];
    const float* Wd       = (const float*)d_in[11];
    const float* bd       = (const float*)d_in[12];
    const float* W1       = (const float*)d_in[13];
    const float* b1       = (const float*)d_in[14];
    const float* W2       = (const float*)d_in[15];
    const float* b2       = (const float*)d_in[16];
    const float* Wp       = (const float*)d_in[17];
    const float* bp       = (const float*)d_in[18];
    const float* Wo       = (const float*)d_in[19];
    const float* bo       = (const float*)d_in[20];
    const float* ngam     = (const float*)d_in[21];
    const float* nbet     = (const float*)d_in[22];
    float* logits = (float*)d_out;

    uint8_t* ws = (uint8_t*)d_ws;
    size_t off = 0;
    auto carve = [&](size_t bytes) -> uint8_t* {
        uint8_t* p = ws + off;
        off = (off + bytes + 255) & ~(size_t)255;
        return p;
    };
    __bf16* emb_b  = (__bf16*)carve((size_t)NROWS * EDIM * 2);
    __bf16* wcat_t = (__bf16*)carve((size_t)NGATES * EDIM * 2);
    __bf16* wc_t   = (__bf16*)carve((size_t)HHEADS * DDIM * DDIM * 2);
    __bf16* w1_t   = (__bf16*)carve((size_t)512 * 128 * 2);
    __bf16* w2_t   = (__bf16*)carve((size_t)128 * 512 * 2);
    __bf16* wp_t   = (__bf16*)carve((size_t)128 * 128 * 2);
    __bf16* wo_t   = (__bf16*)carve((size_t)VOCAB * 128 * 2);
    float*  a_buf  = (float*) carve((size_t)NROWS * 1024 * 4);  // aliased: cout
    float*  b_buf  = (float*) carve((size_t)NROWS * 1024 * 4);  // aliased: g1
    float*  dxt    = (float*) carve((size_t)NROWS * 128 * 4);
    __bf16* hs_b   = (__bf16*)carve((size_t)NROWS * 1024 * 2);
    float*  u_f    = (float*) carve((size_t)NROWS * 128 * 4);
    __bf16* u_b    = (__bf16*)carve((size_t)NROWS * 128 * 2);
    __bf16* hout_b = (__bf16*)carve((size_t)NROWS * 128 * 2);
    __bf16* hp_b   = (__bf16*)carve((size_t)NROWS * 128 * 2);
    float*  coutp  = a_buf;
    __bf16* g1_b   = (__bf16*)b_buf;

    // ---- weight prep ----
    build_gates_w<<<(NGATES * EDIM + 255) / 256, 256, 0, stream>>>(Wa, Wb, Wd, wcat_t);
    transpose_conv<<<dim3((DDIM * DDIM + 255) / 256, HHEADS), 256, 0, stream>>>(Wc, wc_t, DDIM, DDIM);
    transpose_conv<<<dim3((128 * 512 + 255) / 256, 1), 256, 0, stream>>>(W1, w1_t, 128, 512);
    transpose_conv<<<dim3((512 * 128 + 255) / 256, 1), 256, 0, stream>>>(W2, w2_t, 512, 128);
    transpose_conv<<<dim3((128 * 128 + 255) / 256, 1), 256, 0, stream>>>(Wp, wp_t, 128, 128);
    transpose_conv<<<dim3((128 * VOCAB + 255) / 256, 1), 256, 0, stream>>>(Wo, wo_t, 128, VOCAB);

    // ---- embedding ----
    embed_k<<<(NROWS * (EDIM / 8) + 255) / 256, 256, 0, stream>>>(x, emb_tab, emb_b);

    // ---- gates + dxt GEMM ----
    {
        GemmP p = {};
        p.A = emb_b;  p.lda = EDIM;
        p.Bt = wcat_t; p.ldb = EDIM;
        p.ga = a_buf; p.gb = b_buf; p.gd = dxt;
        p.ba = ba; p.bbv = bbv; p.bd = bd;
        p.M = NROWS; p.N = NGATES; p.K = EDIM;
        gemm_wmma<EPI_GATES><<<gemm_blocks(p.M, p.N), 256, 0, stream>>>(p);
    }

    // ---- sequential scan ----
    scan_k<<<16, 256, 0, stream>>>(a_buf, b_buf, hs_b);

    // ---- per-head projection: cout = hs @ Wc[h] + bc[h] + dxt ----
    for (int h = 0; h < HHEADS; ++h) {
        GemmP p = {};
        p.A = hs_b + h * DDIM; p.lda = 1024;
        p.Bt = wc_t + (size_t)h * DDIM * DDIM; p.ldb = DDIM;
        p.bias = bc + h * DDIM;
        p.outf = coutp + h * DDIM; p.ldc = 1024;
        p.res = dxt; p.ldres = DDIM;
        p.M = NROWS; p.N = DDIM; p.K = DDIM;
        gemm_wmma<EPI_RES_F32><<<gemm_blocks(p.M, p.N), 256, 0, stream>>>(p);
    }

    // ---- head LN + combine + z-LN -> u ----
    fuse_z_u<<<(NROWS * 32 + 255) / 256, 256, 0, stream>>>(
        coutp, dxt, hw, hg, hbeta, ngam, nbet, u_f, u_b);

    // ---- MLP: g1 = gelu(u @ W1 + b1) ----
    {
        GemmP p = {};
        p.A = u_b; p.lda = DDIM;
        p.Bt = w1_t; p.ldb = DDIM;
        p.bias = b1;
        p.outb = g1_b; p.ldc = 512;
        p.M = NROWS; p.N = 512; p.K = DDIM;
        gemm_wmma<EPI_GELU_B16><<<gemm_blocks(p.M, p.N), 256, 0, stream>>>(p);
    }
    // ---- h_out = u + (g1 @ W2 + b2) ----
    {
        GemmP p = {};
        p.A = g1_b; p.lda = 512;
        p.Bt = w2_t; p.ldb = 512;
        p.bias = b2;
        p.outb = hout_b; p.ldc = DDIM;
        p.res = u_f; p.ldres = DDIM;
        p.M = NROWS; p.N = DDIM; p.K = 512;
        gemm_wmma<EPI_RES_B16><<<gemm_blocks(p.M, p.N), 256, 0, stream>>>(p);
    }
    // ---- hp = h_out @ Wp + bp ----
    {
        GemmP p = {};
        p.A = hout_b; p.lda = DDIM;
        p.Bt = wp_t; p.ldb = DDIM;
        p.bias = bp;
        p.outb = hp_b; p.ldc = DDIM;
        p.M = NROWS; p.N = DDIM; p.K = DDIM;
        gemm_wmma<EPI_B16><<<gemm_blocks(p.M, p.N), 256, 0, stream>>>(p);
    }
    // ---- logits = hp @ Wo + bo ----
    {
        GemmP p = {};
        p.A = hp_b; p.lda = DDIM;
        p.Bt = wo_t; p.ldb = DDIM;
        p.bias = bo;
        p.outf = logits; p.ldc = VOCAB;
        p.M = NROWS; p.N = VOCAB; p.K = DDIM;
        gemm_wmma<EPI_F32><<<gemm_blocks(p.M, p.N), 256, 0, stream>>>(p);
    }
}
